// GroupedQueryAttention_19456201851387
// MI455X (gfx1250) — compile-verified, matrix operand order checked by reading, not run on hardware
//
#include <hip/hip_runtime.h>

// ---------------------------------------------------------------------------
// GQA attention layer for gfx1250 (MI455X), bf16 WMMA + fp32 accumulate.
// B=2, S=2048, D_IN=2048, H=16, G=4, HD=128, causal.
// Async global->LDS tile staging (ASYNCcnt double buffering) + templated
// GEMM dims for immediate-offset epilogue stores.
// ---------------------------------------------------------------------------

#define BB   2
#define SS   2048
#define DIN  2048
#define HH   16
#define GG   4
#define HD   128
#define GSZ  (HH / GG)          // 4 heads per kv group
#define QSCALE 0.08838834764831845f   // 1/sqrt(128)

typedef float  v8f  __attribute__((ext_vector_type(8)));
typedef __bf16 v16bf __attribute__((ext_vector_type(16)));
typedef __bf16 v8bf  __attribute__((ext_vector_type(8)));
typedef int    v4i  __attribute__((ext_vector_type(4)));

typedef __attribute__((address_space(1))) v4i* gptr4;   // global int4*
typedef __attribute__((address_space(3))) v4i* lptr4;   // LDS int4*

__device__ __forceinline__ v16bf join16(v8bf lo, v8bf hi) {
  v16bf r;
#pragma unroll
  for (int i = 0; i < 8; ++i) { r[i] = lo[i]; r[i + 8] = hi[i]; }
  return r;
}

// 16-byte global -> LDS async copy (GLOBAL_LOAD_ASYNC_TO_LDS_B128, ASYNCcnt)
__device__ __forceinline__ void async_copy16(const __bf16* g, __bf16* l) {
  __builtin_amdgcn_global_load_async_to_lds_b128(
      (gptr4)(g), (lptr4)(l), 0, 0);
}

// ---------------------------------------------------------------------------
// fp32 -> bf16 elementwise convert (grid-stride)
// ---------------------------------------------------------------------------
__global__ __launch_bounds__(256) void cvt_f32_bf16(const float* __restrict__ in,
                                                    __bf16* __restrict__ out,
                                                    long n) {
  long i = (long)blockIdx.x * blockDim.x + threadIdx.x;
  long stride = (long)gridDim.x * blockDim.x;
  for (; i < n; i += stride) out[i] = (__bf16)in[i];
}

// ---------------------------------------------------------------------------
// Tiled bf16 GEMM:  C[m,n] = sum_k A[m,k] * W[n,k]   (both K-contiguous)
// Block tile 128x128, K-tile 32, 256 threads = 8 waves (4 along M x 2 along N).
// Each wave: 2x4 grid of 16x16 WMMA accumulators. Double-buffered async
// global->LDS staging; compile-time N,K so epilogue stores use imm offsets.
// ---------------------------------------------------------------------------
#define LDSW 40   // bf16 elems per LDS row (32 + 8 pad, keeps 16B alignment)

template <int N, int K>
__global__ __launch_bounds__(256)
void gemm_bf16_nt(const __bf16* __restrict__ A, const __bf16* __restrict__ W,
                  float* __restrict__ C, int M) {
  __shared__ __bf16 As[2][128 * LDSW];
  __shared__ __bf16 Ws[2][128 * LDSW];

  const int tid  = threadIdx.x;
  const int lane = tid & 31;
  const int wave = tid >> 5;
  const int wm   = wave & 3;      // 4 waves along M, 32 rows each
  const int wn   = wave >> 2;     // 2 waves along N, 64 cols each
  const int mBase = blockIdx.y * 128;
  const int nBase = blockIdx.x * 128;

  const int fr   = lane & 15;     // fragment row (M for A, N for W)
  const int half = lane >> 4;     // lane half selects K sub-chunk

  // cooperative tile staging: 256 threads x 16 bf16 = 128 rows x 32 cols
  const int ldRow = tid >> 1;
  const int ldCol = (tid & 1) * 16;
  const __bf16* aTile = A + (long)(mBase + ldRow) * K + ldCol;
  const __bf16* wTile = W + (long)(nBase + ldRow) * K + ldCol;

  v8f acc[2][4] = {};

  // prologue: stage K-tile 0 into buffer 0 (4 async b128 per thread)
  {
    __bf16* aD = &As[0][ldRow * LDSW + ldCol];
    __bf16* wD = &Ws[0][ldRow * LDSW + ldCol];
    async_copy16(aTile,     aD);
    async_copy16(aTile + 8, aD + 8);
    async_copy16(wTile,     wD);
    async_copy16(wTile + 8, wD + 8);
  }

  for (int k0 = 0; k0 < K; k0 += 32) {
    const int buf = (k0 >> 5) & 1;
    if (k0 + 32 < K) {            // stage next tile into other buffer
      const __bf16* aS = aTile + k0 + 32;
      const __bf16* wS = wTile + k0 + 32;
      __bf16* aD = &As[buf ^ 1][ldRow * LDSW + ldCol];
      __bf16* wD = &Ws[buf ^ 1][ldRow * LDSW + ldCol];
      async_copy16(aS,     aD);
      async_copy16(aS + 8, aD + 8);
      async_copy16(wS,     wD);
      async_copy16(wS + 8, wD + 8);
      __builtin_amdgcn_s_wait_asynccnt(4);   // current tile landed in LDS
    } else {
      __builtin_amdgcn_s_wait_asynccnt(0);
    }
    __syncthreads();

    v16bf afrag[2], wfrag[4];
#pragma unroll
    for (int i = 0; i < 2; ++i) {
      const __bf16* p = &As[buf][(wm * 32 + i * 16 + fr) * LDSW + half * 8];
      afrag[i] = join16(*(const v8bf*)p, *(const v8bf*)(p + 16));
    }
#pragma unroll
    for (int j = 0; j < 4; ++j) {
      const __bf16* p = &Ws[buf][(wn * 64 + j * 16 + fr) * LDSW + half * 8];
      wfrag[j] = join16(*(const v8bf*)p, *(const v8bf*)(p + 16));
    }
#pragma unroll
    for (int i = 0; i < 2; ++i)
#pragma unroll
      for (int j = 0; j < 4; ++j)
        acc[i][j] = __builtin_amdgcn_wmma_f32_16x16x32_bf16(
            false, afrag[i], false, wfrag[j], (short)0, acc[i][j], false, false);
    __syncthreads();   // protect the buffer we compute from next iteration
  }

  // epilogue: single base pointer, all 64 stores at compile-time offsets
  float* cBase = C + (long)(mBase + wm * 32 + 8 * half) * N + (nBase + wn * 64 + fr);
#pragma unroll
  for (int i = 0; i < 2; ++i)
#pragma unroll
    for (int j = 0; j < 4; ++j)
#pragma unroll
      for (int r = 0; r < 8; ++r)
        cBase[(long)(i * 16 + r) * N + j * 16] = acc[i][j][r];
}

// ---------------------------------------------------------------------------
// Fused RMSNorm + RoPE (+ q scaling) + layout change to bf16.
// One wave per (b, s, slot): slot 0..15 = Q head, 16..19 = K group,
// 20..23 = V group. Lane holds hd = lane + 32*j, j=0..3 (hd+64 is j+2 -> no
// cross-lane traffic for the rotate).
// Qn: [b,h,s,hd]  Kn: [b,g,s,hd]  Vt: [b,g,hd,s] (transposed for P*V WMMA B)
// ---------------------------------------------------------------------------
__global__ __launch_bounds__(128)
void norm_rope_kernel(const float* __restrict__ Qraw, const float* __restrict__ Kraw,
                      const float* __restrict__ Vraw,
                      const float* __restrict__ cosT, const float* __restrict__ sinT,
                      const float* __restrict__ qw, const float* __restrict__ kw,
                      __bf16* __restrict__ Qn, __bf16* __restrict__ Kn,
                      __bf16* __restrict__ Vt) {
  const long gtid = (long)blockIdx.x * blockDim.x + threadIdx.x;
  const int lane = threadIdx.x & 31;
  const long wid = gtid >> 5;
  const int slot = (int)(wid % 24);
  const int s    = (int)((wid / 24) % SS);
  const int b    = (int)(wid / (24L * SS));

  float val[4];
  if (slot < 16) {                       // ---- Q head ----
    const int h = slot;
    const float* src = Qraw + ((long)(b * SS + s)) * (HH * HD) + h * HD;
#pragma unroll
    for (int j = 0; j < 4; ++j) val[j] = src[lane + 32 * j];
    float ss = 0.f;
#pragma unroll
    for (int j = 0; j < 4; ++j) ss += val[j] * val[j];
#pragma unroll
    for (int d = 1; d < 32; d <<= 1) ss += __shfl_xor(ss, d, 32);
    const float r = rsqrtf(ss * (1.0f / HD) + 1e-6f);
    __bf16* dst = Qn + (((long)(b * HH + h)) * SS + s) * HD;
#pragma unroll
    for (int j = 0; j < 4; ++j) {
      const int hd = lane + 32 * j;
      const float nv  = val[j] * r * qw[hd];
      const float rot = (j < 2) ? -(val[j + 2] * r * qw[lane + 32 * (j + 2)])
                                :  (val[j - 2] * r * qw[lane + 32 * (j - 2)]);
      const float o = (nv * cosT[s * HD + hd] + rot * sinT[s * HD + hd]) * QSCALE;
      dst[hd] = (__bf16)o;
    }
  } else if (slot < 20) {                // ---- K group ----
    const int g = slot - 16;
    const float* src = Kraw + ((long)(b * SS + s)) * (GG * HD) + g * HD;
#pragma unroll
    for (int j = 0; j < 4; ++j) val[j] = src[lane + 32 * j];
    float ss = 0.f;
#pragma unroll
    for (int j = 0; j < 4; ++j) ss += val[j] * val[j];
#pragma unroll
    for (int d = 1; d < 32; d <<= 1) ss += __shfl_xor(ss, d, 32);
    const float r = rsqrtf(ss * (1.0f / HD) + 1e-6f);
    __bf16* dst = Kn + (((long)(b * GG + g)) * SS + s) * HD;
#pragma unroll
    for (int j = 0; j < 4; ++j) {
      const int hd = lane + 32 * j;
      const float nv  = val[j] * r * kw[hd];
      const float rot = (j < 2) ? -(val[j + 2] * r * kw[lane + 32 * (j + 2)])
                                :  (val[j - 2] * r * kw[lane + 32 * (j - 2)]);
      dst[hd] = (__bf16)(nv * cosT[s * HD + hd] + rot * sinT[s * HD + hd]);
    }
  } else {                               // ---- V group (transpose store) ----
    const int g = slot - 20;
    const float* src = Vraw + ((long)(b * SS + s)) * (GG * HD) + g * HD;
    __bf16* dst = Vt + ((long)(b * GG + g)) * HD * SS;
#pragma unroll
    for (int j = 0; j < 4; ++j) {
      const int hd = lane + 32 * j;
      dst[(long)hd * SS + s] = (__bf16)src[hd];
    }
  }
}

// ---------------------------------------------------------------------------
// Flash attention: block = 4 waves, each wave owns 16 query rows of one head.
// kv tiles of 32; QK^T and P*V via bf16 WMMA, online softmax in fp32.
// ---------------------------------------------------------------------------
__global__ __launch_bounds__(128)
void attn_kernel(const __bf16* __restrict__ Qn, const __bf16* __restrict__ Kn,
                 const __bf16* __restrict__ Vt, __bf16* __restrict__ ctx) {
  const int b = blockIdx.z, h = blockIdx.y;
  const int g = h / GSZ;
  const int wave = threadIdx.x >> 5, lane = threadIdx.x & 31;
  const int fr = lane & 15, half = lane >> 4;
  const int qBase = blockIdx.x * 64 + wave * 16;

  __shared__ __bf16 Plds[4][16 * LDSW];
  __bf16* P = &Plds[wave][0];

  // Q fragments (16 rows x 128, 4 K-chunks of 32)
  const __bf16* qPtr = Qn + (((long)(b * HH + h)) * SS + (qBase + fr)) * HD;
  v16bf qf[4];
#pragma unroll
  for (int c = 0; c < 4; ++c) {
    const __bf16* p = qPtr + c * 32 + half * 8;
    qf[c] = join16(*(const v8bf*)p, *(const v8bf*)(p + 16));
  }

  float mrow[8], lrow[8];
#pragma unroll
  for (int r = 0; r < 8; ++r) { mrow[r] = -3.0e38f; lrow[r] = 0.f; }
  v8f o[8] = {};

  const __bf16* kBase = Kn + ((long)(b * GG + g)) * SS * HD;
  const __bf16* vBase = Vt + ((long)(b * GG + g)) * HD * SS;
  const int qMax = qBase + 15;

  for (int kb = 0; kb <= qMax; kb += 32) {
    // ---- scores: 16 q-rows x 32 kv-cols ----
    v8f sc[2] = {};
#pragma unroll
    for (int sub = 0; sub < 2; ++sub) {
      const __bf16* kPtr = kBase + (long)(kb + sub * 16 + fr) * HD;
#pragma unroll
      for (int c = 0; c < 4; ++c) {
        const __bf16* p = kPtr + c * 32 + half * 8;
        v16bf kf = join16(*(const v8bf*)p, *(const v8bf*)(p + 16));
        sc[sub] = __builtin_amdgcn_wmma_f32_16x16x32_bf16(
            false, qf[c], false, kf, (short)0, sc[sub], false, false);
      }
    }
    // ---- causal mask + online softmax ----
    float alpha[8];
#pragma unroll
    for (int r = 0; r < 8; ++r) {
      const int q = qBase + r + 8 * half;
      float s0 = (kb + fr      <= q) ? sc[0][r] : -3.0e38f;
      float s1 = (kb + 16 + fr <= q) ? sc[1][r] : -3.0e38f;
      float mx = fmaxf(s0, s1);
#pragma unroll
      for (int d = 1; d < 16; d <<= 1) mx = fmaxf(mx, __shfl_xor(mx, d, 32));
      const float mNew = fmaxf(mrow[r], mx);
      alpha[r] = __expf(mrow[r] - mNew);
      mrow[r] = mNew;
      const float p0 = __expf(s0 - mNew);
      const float p1 = __expf(s1 - mNew);
      sc[0][r] = p0; sc[1][r] = p1;
      float ps = p0 + p1;
#pragma unroll
      for (int d = 1; d < 16; d <<= 1) ps += __shfl_xor(ps, d, 32);
      lrow[r] = lrow[r] * alpha[r] + ps;
    }
#pragma unroll
    for (int t = 0; t < 8; ++t)
#pragma unroll
      for (int r = 0; r < 8; ++r) o[t][r] *= alpha[r];

    // ---- re-shape P (C layout) -> A fragment via LDS ----
#pragma unroll
    for (int sub = 0; sub < 2; ++sub)
#pragma unroll
      for (int r = 0; r < 8; ++r)
        P[(r + 8 * half) * LDSW + sub * 16 + fr] = (__bf16)sc[sub][r];
    asm volatile("s_wait_dscnt 0" ::: "memory");
    const __bf16* pp = &P[fr * LDSW + half * 8];
    v16bf pf = join16(*(const v8bf*)pp, *(const v8bf*)(pp + 16));

    // ---- P x V : 8 hd subtiles of 16 ----
#pragma unroll
    for (int t = 0; t < 8; ++t) {
      const __bf16* vPtr = vBase + (long)(t * 16 + fr) * SS + kb + half * 8;
      v16bf vf = join16(*(const v8bf*)vPtr, *(const v8bf*)(vPtr + 16));
      o[t] = __builtin_amdgcn_wmma_f32_16x16x32_bf16(
          false, pf, false, vf, (short)0, o[t], false, false);
    }
  }

  // ---- normalize + store ctx[b, s, h*HD + hd] as bf16 ----
#pragma unroll
  for (int r = 0; r < 8; ++r) lrow[r] = 1.0f / lrow[r];
#pragma unroll
  for (int t = 0; t < 8; ++t)
#pragma unroll
    for (int r = 0; r < 8; ++r) {
      const int s = qBase + r + 8 * half;
      const int e = h * HD + t * 16 + fr;
      ctx[((long)(b * SS + s)) * (HH * HD) + e] = (__bf16)(o[t][r] * lrow[r]);
    }
}

// ---------------------------------------------------------------------------
// Host orchestration
// ---------------------------------------------------------------------------
static inline size_t alignUp(size_t x, size_t a) { return (x + a - 1) & ~(a - 1); }

extern "C" void kernel_launch(void* const* d_in, const int* in_sizes, int n_in,
                              void* d_out, int out_size, void* d_ws, size_t ws_size,
                              hipStream_t stream) {
  const float* x   = (const float*)d_in[0];
  // d_in[1] = mask (unused; causal computed analytically)
  const float* cosT = (const float*)d_in[2];
  const float* sinT = (const float*)d_in[3];
  const float* Wq  = (const float*)d_in[4];
  const float* Wk  = (const float*)d_in[5];
  const float* Wv  = (const float*)d_in[6];
  const float* Wo  = (const float*)d_in[7];
  const float* qw  = (const float*)d_in[8];
  const float* kw  = (const float*)d_in[9];
  float* out = (float*)d_out;

  const long M = (long)BB * SS;          // 4096 token rows
  char* ws = (char*)d_ws;
  size_t off = 0;
  auto carve = [&](size_t bytes) { char* p = ws + off; off = alignUp(off + bytes, 256); return p; };

  __bf16* xb   = (__bf16*)carve(M * DIN * 2);
  __bf16* Wqb  = (__bf16*)carve((size_t)HH * HD * DIN * 2);
  __bf16* Wkb  = (__bf16*)carve((size_t)GG * HD * DIN * 2);
  __bf16* Wvb  = (__bf16*)carve((size_t)GG * HD * DIN * 2);
  __bf16* Wob  = (__bf16*)carve((size_t)DIN * HH * HD * 2);
  float*  Qraw = (float*)carve(M * HH * HD * 4);
  float*  Kraw = (float*)carve(M * GG * HD * 4);
  float*  Vraw = (float*)carve(M * GG * HD * 4);
  __bf16* Qn   = (__bf16*)carve(M * HH * HD * 2);
  __bf16* Kn   = (__bf16*)carve(M * GG * HD * 2);
  __bf16* Vt   = (__bf16*)carve(M * GG * HD * 2);
  __bf16* ctx  = (__bf16*)carve(M * HH * HD * 2);
  (void)ws_size;

  // 1) fp32 -> bf16 converts
  {
    long n;
    n = M * DIN;                cvt_f32_bf16<<<2048, 256, 0, stream>>>(x,  xb,  n);
    n = (long)HH * HD * DIN;    cvt_f32_bf16<<<2048, 256, 0, stream>>>(Wq, Wqb, n);
    n = (long)GG * HD * DIN;    cvt_f32_bf16<<<1024, 256, 0, stream>>>(Wk, Wkb, n);
    n = (long)GG * HD * DIN;    cvt_f32_bf16<<<1024, 256, 0, stream>>>(Wv, Wvb, n);
    n = (long)DIN * HH * HD;    cvt_f32_bf16<<<2048, 256, 0, stream>>>(Wo, Wob, n);
  }

  // 2) QKV projections (bf16 WMMA, fp32 out)
  gemm_bf16_nt<HH * HD, DIN><<<dim3((HH * HD) / 128, M / 128), 256, 0, stream>>>(xb, Wqb, Qraw, M);
  gemm_bf16_nt<GG * HD, DIN><<<dim3((GG * HD) / 128, M / 128), 256, 0, stream>>>(xb, Wkb, Kraw, M);
  gemm_bf16_nt<GG * HD, DIN><<<dim3((GG * HD) / 128, M / 128), 256, 0, stream>>>(xb, Wvb, Vraw, M);

  // 3) RMSNorm + RoPE + scaling + layout (wave per (b,s,slot), 24 slots)
  {
    const long waves = (long)BB * SS * 24;
    const int blocks = (int)(waves * 32 / 128);
    norm_rope_kernel<<<blocks, 128, 0, stream>>>(Qraw, Kraw, Vraw, cosT, sinT,
                                                 qw, kw, Qn, Kn, Vt);
  }

  // 4) flash attention (wave per 16 query rows)
  attn_kernel<<<dim3(SS / 64, HH, BB), 128, 0, stream>>>(Qn, Kn, Vt, ctx);

  // 5) output projection -> fp32 d_out
  gemm_bf16_nt<DIN, HH * HD><<<dim3(DIN / 128, M / 128), 256, 0, stream>>>(ctx, Wob, out, M);
  (void)out_size; (void)n_in; (void)in_sizes;
}
